// Model_86646670229685
// MI455X (gfx1250) — matrix operand order repaired and checked by reading, not compiled, not run on hardware
//
#include <hip/hip_runtime.h>
#include <cmath>
#include <cstring>

#define B_    8
#define TRAW_ 4096
#define CIN_  8
#define D_    768
#define H_    4
#define FF_   3072
#define L_    6
#define DQ_   192
#define V_    40
#define TT_   128
#define TS_   512
#define RW_   201

typedef __attribute__((ext_vector_type(16))) _Float16 v16h;
typedef __attribute__((ext_vector_type(8)))  _Float16 v8h;
typedef __attribute__((ext_vector_type(8)))  float    v8f;

// ---------------------------------------------------------------------------
// Generic batched WMMA GEMM:  C[M,N] = act( alpha_or_scaleN * (A[M,K] @ B^T) + bias )
// A: f16 row-major [M,K] (lda), B: f16 packed [N,K] (ldb).
// Register-blocked: each wave owns a 64x32 tile = 4x2 subtiles of 16x16,
// so one K-step (32) = 12 b128 loads feeding 8 v_wmma (A frags reused across
// N-subtiles, B frags across M-subtiles -> ~21 FLOP/byte vs 8 for 1x1).
// Dual output: optional f32 C and optional f16 Ch (each with own strides).
// Batch z = blockIdx.y decomposed as (zo,zi) = (z/innerCnt, z%innerCnt).
// ---------------------------------------------------------------------------
#define MT_ 4
#define NT_ 2

struct GemmP {
  const _Float16* A; const _Float16* B; float* C; _Float16* Ch;
  const float* bias; const float* scaleN;
  long aO, aI, bO, bI, cO, cI, hO, hI;
  int lda, ldb, ldc, ldch;
  int M, N, K, batches, innerCnt, relu;
  float alpha;
};

__global__ __launch_bounds__(256) void k_gemm(GemmP p) {
  const int wave = threadIdx.x >> 5;
  const int lane = threadIdx.x & 31;
  const int tilesN = (p.N + 16 * NT_ - 1) / (16 * NT_);
  const int tilesM = (p.M + 16 * MT_ - 1) / (16 * MT_);
  const long tile = (long)blockIdx.x * 8 + wave;
  if (tile >= (long)tilesM * tilesN) return;          // wave-uniform: EXEC stays all-ones
  const int z  = blockIdx.y;
  const int zo = z / p.innerCnt, zi = z % p.innerCnt;
  const int tm = (int)(tile / tilesN), tn = (int)(tile % tilesN);
  const int m0 = tm * 16 * MT_, n0 = tn * 16 * NT_;
  const int hl = lane >> 4, l15 = lane & 15;
  const _Float16* Ab = p.A + zo * p.aO + zi * p.aI;
  const _Float16* Bb = p.B + zo * p.bO + zi * p.bI;
  const _Float16* arow[MT_];
  const _Float16* brow[NT_];
#pragma unroll
  for (int i = 0; i < MT_; ++i) {                     // clamp: loads stay uniform-legal
    int am = m0 + i * 16 + l15; if (am >= p.M) am = p.M - 1;
    arow[i] = Ab + (long)am * p.lda;
  }
#pragma unroll
  for (int j = 0; j < NT_; ++j) {
    int bn = n0 + j * 16 + l15; if (bn >= p.N) bn = p.N - 1;
    brow[j] = Bb + (long)bn * p.ldb;
  }
  v8f acc[MT_][NT_] = {};
  for (int kb = 0; kb < p.K; kb += 32) {
    v16h af[MT_], bf[NT_];
#pragma unroll
    for (int i = 0; i < MT_; ++i) {
      // A 16x32 f16 fragment: lanes 0-15 hold K 0-15, lanes 16-31 hold K 16-31
      union { v16h v; v8h h[2]; } u;
      u.h[0] = *(const v8h*)(arow[i] + kb + hl * 16);
      u.h[1] = *(const v8h*)(arow[i] + kb + hl * 16 + 8);
      af[i] = u.v;
      __builtin_prefetch(arow[i] + kb + 64, 0, 1);
    }
#pragma unroll
    for (int j = 0; j < NT_; ++j) {
      // B 32x16 f16 fragment: lane group hl holds K kb+hl*16 .. +15 of column l15
      bf[j] = *(const v16h*)(brow[j] + kb + hl * 16);
      __builtin_prefetch(brow[j] + kb + 64, 0, 1);
    }
#pragma unroll
    for (int i = 0; i < MT_; ++i)
#pragma unroll
      for (int j = 0; j < NT_; ++j)
        acc[i][j] = __builtin_amdgcn_wmma_f32_16x16x32_f16(false, af[i], false, bf[j],
                                                           (short)0, acc[i][j], false, false);
  }
  const long cb = zo * p.cO + zi * p.cI;
  const long hb = zo * p.hO + zi * p.hI;
#pragma unroll
  for (int j = 0; j < NT_; ++j) {
    const int col  = n0 + j * 16 + l15;
    const int cidx = (col < p.N) ? col : p.N - 1;
    const float sc = p.scaleN ? p.scaleN[cidx] : p.alpha;
    const float bs = p.bias ? p.bias[cidx] : 0.f;
#pragma unroll
    for (int i = 0; i < MT_; ++i) {
#pragma unroll
      for (int r = 0; r < 8; ++r) {
        const int row = m0 + i * 16 + hl * 8 + r;     // C layout: VGPR r -> row (lane/16)*8+r
        if (row < p.M && col < p.N) {
          float val = acc[i][j][r] * sc + bs;
          if (p.relu) val = fmaxf(val, 0.f);
          if (p.C)  p.C [cb + (long)row * p.ldc  + col] = val;
          if (p.Ch) p.Ch[hb + (long)row * p.ldch + col] = (_Float16)val;
        }
      }
    }
  }
}

// --------------------------- packing / small kernels ------------------------
// f32 [batches,R,C] -> f16 [batches,C,R]  (weight transpose into [N,K] form)
__global__ void k_packT(const float* in, _Float16* out, int R, int C, int batches) {
  long idx = (long)blockIdx.x * 256 + threadIdx.x;
  long rc = (long)R * C;
  if (idx >= rc * batches) return;
  long b = idx / rc, rem = idx % rc;
  int c = (int)(rem / R), r = (int)(rem % R);
  out[idx] = (_Float16)in[b * rc + (long)r * C + c];
}

// f32 [rows,Kreal] -> f16 [rows,Kpad] zero-padded (conv weights / plain copies)
__global__ void k_packW(const float* in, _Float16* out, int rows, int Kreal, int Kpad) {
  long idx = (long)blockIdx.x * 256 + threadIdx.x;
  if (idx >= (long)rows * Kpad) return;
  int k = (int)(idx % Kpad); int r = (int)(idx / Kpad);
  out[idx] = (k < Kreal) ? (_Float16)in[(long)r * Kreal + k] : (_Float16)0.f;
}

// f16 [B,T,H,DQ] -> f16 [B,H,DQ,T]  (V transposed for probs@V GEMM)
__global__ void k_packvt(const _Float16* in, _Float16* out, int T) {
  long idx = (long)blockIdx.x * 256 + threadIdx.x;
  if (idx >= (long)B_ * H_ * DQ_ * T) return;
  int t = (int)(idx % T); long r = idx / T;
  int a = (int)(r % DQ_); r /= DQ_;
  int h = (int)(r % H_);  int b = (int)(r / H_);
  out[idx] = in[(((long)b * T + t) * H_ + h) * DQ_ + a];
}

// im2col with K padded to Kpad; activations kept [B,T,C] f32
__global__ void k_im2col(const float* in, _Float16* out, int Cin, int Tin, int Tout,
                         int ks, int stride, int Kpad) {
  long idx = (long)blockIdx.x * 256 + threadIdx.x;
  if (idx >= (long)B_ * Tout * Kpad) return;
  int kc = (int)(idx % Kpad);
  int t  = (int)((idx / Kpad) % Tout);
  int b  = (int)(idx / ((long)Kpad * Tout));
  float v = 0.f;
  if (kc < Cin * ks) {
    int ci = kc / ks, kk = kc % ks;
    int ti = t * stride + kk - (ks - 1) / 2;
    if (ti >= 0 && ti < Tin) v = in[((long)b * Tin + ti) * Cin + ci];
  }
  out[idx] = (_Float16)v;
}

__global__ void k_bnfold(const float* cb, const float* g, const float* bb,
                         const float* m, const float* v, float* scale, float* bias2) {
  int i = blockIdx.x * 256 + threadIdx.x;
  if (i >= D_) return;
  float s = g[i] * rsqrtf(v[i] + 1e-5f);
  scale[i] = s;
  bias2[i] = (cb[i] - m[i]) * s + bb[i];
}

__global__ void k_addrelu(const float* a, const float* b, float* out, _Float16* outh, long n) {
  long i = (long)blockIdx.x * 256 + threadIdx.x;
  if (i >= n) return;
  float r = fmaxf(a[i] + b[i], 0.f);
  out[i] = r;
  if (outh) outh[i] = (_Float16)r;
}

__global__ void k_srcpad(const float* x, int* pad) {
  int i = blockIdx.x * 256 + threadIdx.x;
  if (i < B_ * TS_) pad[i] = (x[(long)i * D_] == 0.f) ? 1 : 0;
}
__global__ void k_tgtpad(const int* y, int* pad) {
  int i = blockIdx.x * 256 + threadIdx.x;
  if (i < B_ * TT_) pad[i] = (y[i] == 0) ? 1 : 0;
}

// token embedding + interleaved sinusoidal PE, dual f32/f16 output
__global__ void k_embed(const int* y, const float* emb, float* out, _Float16* outh) {
  long idx = (long)blockIdx.x * 256 + threadIdx.x;
  if (idx >= (long)B_ * TT_ * D_) return;
  int d = (int)(idx % D_);
  int i = (int)((idx / D_) % TT_);
  int b = (int)(idx / ((long)D_ * TT_));
  int tok = y[b * TT_ + i];
  float e = emb[(long)tok * D_ + d];
  int j2 = (d >> 1) << 1;
  float div = expf((float)j2 * (-9.210340371976184f / (float)D_)); // -ln(10000)/D
  float ang = (float)i * div;
  float pe = (d & 1) ? cosf(ang) : sinf(ang);
  float r = e + pe;
  out[idx] = r; outh[idx] = (_Float16)r;
}

// fused: rel-position gather + causal/pad mask + softmax -> f16 probs
__global__ __launch_bounds__(256) void k_softmax(const float* logits, const float* rl,
                                                 const int* pad, _Float16* probs,
                                                 int T, int S, int causal, int rwin) {
  __shared__ float red[256];
  const int row = blockIdx.x;          // b*H*T + h*T + t
  const int t = row % T;
  const int b = row / (H_ * T);
  const float* lr = logits + (long)row * S;
  float vals[2];
  float mx = -3.0e38f;
#pragma unroll
  for (int j = 0; j < 2; ++j) {
    int s = threadIdx.x + j * 256;
    float v = -3.0e38f;
    if (s < S) {
      v = lr[s];
      if (rl) {
        int dd = (rwin - 1) >> 1;
        int r = s - t; r = r < -dd ? -dd : (r > dd ? dd : r);
        v += rl[(long)row * rwin + r + dd];
      }
      if (causal && s > t) v = -1e9f;
      if (pad && pad[b * S + s]) v = -1e9f;
    }
    vals[j] = v; mx = fmaxf(mx, v);
  }
  red[threadIdx.x] = mx; __syncthreads();
  for (int st = 128; st > 0; st >>= 1) {
    if (threadIdx.x < st) red[threadIdx.x] = fmaxf(red[threadIdx.x], red[threadIdx.x + st]);
    __syncthreads();
  }
  mx = red[0]; __syncthreads();
  float sum = 0.f;
#pragma unroll
  for (int j = 0; j < 2; ++j) {
    int s = threadIdx.x + j * 256;
    float e = (s < S) ? expf(vals[j] - mx) : 0.f;
    vals[j] = e; sum += e;
  }
  red[threadIdx.x] = sum; __syncthreads();
  for (int st = 128; st > 0; st >>= 1) {
    if (threadIdx.x < st) red[threadIdx.x] += red[threadIdx.x + st];
    __syncthreads();
  }
  float inv = 1.f / red[0];
#pragma unroll
  for (int j = 0; j < 2; ++j) {
    int s = threadIdx.x + j * 256;
    if (s < S) probs[(long)row * S + s] = (_Float16)(vals[j] * inv);
  }
}

// fused residual-add + LayerNorm over D=768, dual f32/f16 output (in-place safe)
__global__ __launch_bounds__(256) void k_addln(const float* x, const float* a,
                                               const float* g, const float* bt,
                                               float* out, _Float16* outh) {
  __shared__ float red[256];
  const long row = blockIdx.x;
  const float* xr = x + row * (long)D_;
  const float* ar = a + row * (long)D_;
  float vals[3]; float s = 0.f;
#pragma unroll
  for (int j = 0; j < 3; ++j) {
    int d = threadIdx.x + j * 256;
    float t = xr[d] + ar[d]; vals[j] = t; s += t;
  }
  red[threadIdx.x] = s; __syncthreads();
  for (int st = 128; st > 0; st >>= 1) {
    if (threadIdx.x < st) red[threadIdx.x] += red[threadIdx.x + st];
    __syncthreads();
  }
  float mu = red[0] * (1.f / D_); __syncthreads();
  s = 0.f;
#pragma unroll
  for (int j = 0; j < 3; ++j) { float dd = vals[j] - mu; s += dd * dd; }
  red[threadIdx.x] = s; __syncthreads();
  for (int st = 128; st > 0; st >>= 1) {
    if (threadIdx.x < st) red[threadIdx.x] += red[threadIdx.x + st];
    __syncthreads();
  }
  float inv = rsqrtf(red[0] * (1.f / D_) + 1e-5f);
#pragma unroll
  for (int j = 0; j < 3; ++j) {
    int d = threadIdx.x + j * 256;
    float r = (vals[j] - mu) * inv * g[d] + bt[d];
    out[row * (long)D_ + d]  = r;
    outh[row * (long)D_ + d] = (_Float16)r;
  }
}

// ---------------------------------------------------------------------------
extern "C" void kernel_launch(void* const* d_in, const int* in_sizes, int n_in,
                              void* d_out, int out_size, void* d_ws, size_t ws_size,
                              hipStream_t stream) {
  (void)in_sizes; (void)n_in; (void)out_size; (void)ws_size;
  const float* x_raw = (const float*)d_in[0];
  const int*   y     = (const int*)d_in[1];
  auto F = [&](int i) { return (const float*)d_in[i]; };

  char* ws = (char*)d_ws;
  size_t off = 0;
  auto alloc = [&](size_t bytes) { off = (off + 255) & ~(size_t)255; size_t o = off; off += bytes; return o; };
  auto PF = [&](size_t o) { return (float*)(ws + o); };
  auto PH = [&](size_t o) { return (_Float16*)(ws + o); };
  auto PI = [&](size_t o) { return (int*)(ws + o); };
  auto g1 = [](long n) { return dim3((unsigned)((n + 255) / 256)); };

  // ---- persistent buffers ----
  size_t o_convh  = alloc((size_t)B_ * TS_ * D_ * 2);   // conv output f16 (GEMM A)
  size_t o_xf     = alloc((size_t)B_ * TS_ * D_ * 4);   // encoder state f32
  size_t o_xh     = alloc((size_t)B_ * TS_ * D_ * 2);   // encoder state f16 (=mem)
  size_t o_tf     = alloc((size_t)B_ * TT_ * D_ * 4);
  size_t o_th     = alloc((size_t)B_ * TT_ * D_ * 2);
  size_t o_srcpad = alloc((size_t)B_ * TS_ * 4);
  size_t o_tgtpad = alloc((size_t)B_ * TT_ * 4);
  size_t o_wraw   = alloc((size_t)D_ * D_ * 2);
  size_t o_waux   = alloc((size_t)V_ * D_ * 2);
  size_t o_wout   = alloc((size_t)V_ * D_ * 2);
  size_t scratch0 = (off + 255) & ~(size_t)255;

  // ---- conv-phase overlay ----
  off = scratch0;
  size_t o_bufA  = alloc((size_t)B_ * 2048 * D_ * 4);
  size_t o_bufB  = alloc((size_t)B_ * 2048 * D_ * 4);
  size_t o_bufC  = alloc((size_t)B_ * 2048 * D_ * 4);
  size_t o_col   = alloc((size_t)B_ * 2048 * 2304 * 2);
  size_t o_scale = alloc(D_ * 4);
  size_t o_bias2 = alloc(D_ * 4);
  size_t o_wcv   = alloc((size_t)D_ * 2304 * 2);
  size_t o_wcvr  = alloc((size_t)D_ * D_ * 2);

  // ---- transformer-phase overlay (same region; stream order makes aliasing safe) ----
  off = scratch0;
  size_t o_q      = alloc((size_t)B_ * TS_ * D_ * 2);
  size_t o_k      = alloc((size_t)B_ * TS_ * D_ * 2);
  size_t o_v      = alloc((size_t)B_ * TS_ * D_ * 2);
  size_t o_vt     = alloc((size_t)B_ * TS_ * D_ * 2);
  size_t o_oa     = alloc((size_t)B_ * TS_ * D_ * 2);
  size_t o_logits = alloc((size_t)B_ * H_ * TS_ * TS_ * 4);
  size_t o_rl     = alloc((size_t)B_ * H_ * TS_ * RW_ * 4);
  size_t o_probs  = alloc((size_t)B_ * H_ * TS_ * TS_ * 2);
  size_t o_attn   = alloc((size_t)B_ * TS_ * D_ * 4);
  size_t o_ffh    = alloc((size_t)B_ * TS_ * FF_ * 2);
  size_t o_ffo    = alloc((size_t)B_ * TS_ * D_ * 4);
  size_t o_wq     = alloc((size_t)D_ * D_ * 2);
  size_t o_wk     = alloc((size_t)D_ * D_ * 2);
  size_t o_wv     = alloc((size_t)D_ * D_ * 2);
  size_t o_wo     = alloc((size_t)D_ * D_ * 2);
  size_t o_rel    = alloc((size_t)H_ * RW_ * DQ_ * 2);
  size_t o_wl1    = alloc((size_t)D_ * FF_ * 2);
  size_t o_wl2    = alloc((size_t)D_ * FF_ * 2);

  const float rsq = 1.0f / sqrtf((float)DQ_);

  auto gemmL = [&](const GemmP& p) {
    long tiles = (long)((p.M + 16 * MT_ - 1) / (16 * MT_)) * ((p.N + 16 * NT_ - 1) / (16 * NT_));
    dim3 grid((unsigned)((tiles + 7) / 8), (unsigned)p.batches, 1);
    k_gemm<<<grid, dim3(256), 0, stream>>>(p);
  };
  auto gp0 = []() { GemmP p{}; p.batches = 1; p.innerCnt = 1; p.alpha = 1.f; return p; };
  auto packTL = [&](const float* in, _Float16* out, int R, int C, int batches) {
    k_packT<<<g1((long)R * C * batches), 256, 0, stream>>>(in, out, R, C, batches);
  };
  auto packWL = [&](const float* in, _Float16* out, int rows, int Kreal, int Kpad) {
    k_packW<<<g1((long)rows * Kpad), 256, 0, stream>>>(in, out, rows, Kreal, Kpad);
  };

  // ---- pack global projection weights ([K,N] f32 -> [N,K] f16) ----
  packTL(F(57), PH(o_wraw), D_, D_, 1);   // w_raw_in
  packTL(F(91), PH(o_waux), D_, V_, 1);   // w_aux
  packTL(F(93), PH(o_wout), D_, V_, 1);   // w_out

  // =========================== conv frontend ===============================
  {
    int cin = CIN_, Tin = TRAW_;
    for (int blk = 0; blk < 3; ++blk) {
      int base = 3 + 18 * blk;
      int Tout = Tin / 2;
      int Kmain = cin * 3, KmainP = ((Kmain + 31) / 32) * 32;
      int Kres  = cin,     KresP  = ((Kres + 31) / 32) * 32;
      const float* inp = (blk == 0) ? x_raw : PF(o_bufA);
      // conv1 (k=3, stride 2) + bn1 + relu  -> bufB
      packWL(F(base + 0), PH(o_wcv), D_, Kmain, KmainP);
      k_bnfold<<<g1(D_), 256, 0, stream>>>(F(base + 1), F(base + 6), F(base + 7),
                                           F(base + 8), F(base + 9), PF(o_scale), PF(o_bias2));
      k_im2col<<<g1((long)B_ * Tout * KmainP), 256, 0, stream>>>(inp, PH(o_col), cin, Tin, Tout, 3, 2, KmainP);
      { GemmP p = gp0(); p.A = PH(o_col); p.lda = KmainP; p.B = PH(o_wcv); p.ldb = KmainP;
        p.C = PF(o_bufB); p.ldc = D_; p.M = B_ * Tout; p.N = D_; p.K = KmainP;
        p.scaleN = PF(o_scale); p.bias = PF(o_bias2); p.relu = 1; gemmL(p); }
      // conv2 (k=3, stride 1) + bn2        -> bufC
      packWL(F(base + 2), PH(o_wcv), D_, D_ * 3, D_ * 3);
      k_bnfold<<<g1(D_), 256, 0, stream>>>(F(base + 3), F(base + 10), F(base + 11),
                                           F(base + 12), F(base + 13), PF(o_scale), PF(o_bias2));
      k_im2col<<<g1((long)B_ * Tout * (D_ * 3)), 256, 0, stream>>>(PF(o_bufB), PH(o_col), D_, Tout, Tout, 3, 1, D_ * 3);
      { GemmP p = gp0(); p.A = PH(o_col); p.lda = D_ * 3; p.B = PH(o_wcv); p.ldb = D_ * 3;
        p.C = PF(o_bufC); p.ldc = D_; p.M = B_ * Tout; p.N = D_; p.K = D_ * 3;
        p.scaleN = PF(o_scale); p.bias = PF(o_bias2); gemmL(p); }
      // residual (k=1, stride 2) + rbn     -> bufB
      packWL(F(base + 4), PH(o_wcvr), D_, Kres, KresP);
      k_bnfold<<<g1(D_), 256, 0, stream>>>(F(base + 5), F(base + 14), F(base + 15),
                                           F(base + 16), F(base + 17), PF(o_scale), PF(o_bias2));
      k_im2col<<<g1((long)B_ * Tout * KresP), 256, 0, stream>>>(inp, PH(o_col), cin, Tin, Tout, 1, 2, KresP);
      { GemmP p = gp0(); p.A = PH(o_col); p.lda = KresP; p.B = PH(o_wcvr); p.ldb = KresP;
        p.C = PF(o_bufB); p.ldc = D_; p.M = B_ * Tout; p.N = D_; p.K = KresP;
        p.scaleN = PF(o_scale); p.bias = PF(o_bias2); gemmL(p); }
      // relu(h + r) -> bufA (+ f16 copy on last block)
      k_addrelu<<<g1((long)B_ * Tout * D_), 256, 0, stream>>>(PF(o_bufC), PF(o_bufB), PF(o_bufA),
                                                              (blk == 2) ? PH(o_convh) : nullptr,
                                                              (long)B_ * Tout * D_);
      cin = D_; Tin = Tout;
    }
  }

  // x = conv_out @ w_raw_in + b ; src_pad = (x[:,:,0] == 0)
  { GemmP p = gp0(); p.A = PH(o_convh); p.lda = D_; p.B = PH(o_wraw); p.ldb = D_;
    p.C = PF(o_xf); p.ldc = D_; p.Ch = PH(o_xh); p.ldch = D_;
    p.M = B_ * TS_; p.N = D_; p.K = D_; p.bias = F(58); gemmL(p); }
  k_srcpad<<<g1(B_ * TS_), 256, 0, stream>>>(PF(o_xf), PI(o_srcpad));
  k_tgtpad<<<g1(B_ * TT_), 256, 0, stream>>>(y, PI(o_tgtpad));
  k_embed<<<g1((long)B_ * TT_ * D_), 256, 0, stream>>>(y, F(59), PF(o_tf), PH(o_th));

  // ------- shared attention block (enc self / dec self / dec cross) --------
  auto attn = [&](const _Float16* xq, const _Float16* xkv, int Tq, int Tk,
                  const float* wqf, const float* wkf, const float* wvf, const float* wof,
                  const float* relf, int causal, const int* pad, float* attnout) {
    packTL(wqf, PH(o_wq), D_, DQ_, H_);
    packTL(wkf, PH(o_wk), D_, DQ_, H_);
    packTL(wvf, PH(o_wv), D_, DQ_, H_);
    packTL(wof, PH(o_wo), D_, D_, 1);
    if (relf) packWL(relf, PH(o_rel), H_ * RW_, DQ_, DQ_);
    // q / k / v projections -> f16 [b,t,h,a]
    { GemmP p = gp0(); p.A = xq; p.lda = D_; p.B = PH(o_wq); p.ldb = D_;
      p.Ch = PH(o_q); p.ldch = D_; p.M = B_ * Tq; p.N = D_; p.K = D_; gemmL(p); }
    { GemmP p = gp0(); p.A = xkv; p.lda = D_; p.B = PH(o_wk); p.ldb = D_;
      p.Ch = PH(o_k); p.ldch = D_; p.M = B_ * Tk; p.N = D_; p.K = D_; gemmL(p); }
    { GemmP p = gp0(); p.A = xkv; p.lda = D_; p.B = PH(o_wv); p.ldb = D_;
      p.Ch = PH(o_v); p.ldch = D_; p.M = B_ * Tk; p.N = D_; p.K = D_; gemmL(p); }
    k_packvt<<<g1((long)B_ * H_ * DQ_ * Tk), 256, 0, stream>>>(PH(o_v), PH(o_vt), Tk);
    // logits[b,h,t,s] = (q/sqrt(dq)) . k
    { GemmP p = gp0(); p.batches = B_ * H_; p.innerCnt = H_;
      p.A = PH(o_q); p.aO = (long)Tq * D_; p.aI = DQ_; p.lda = D_;
      p.B = PH(o_k); p.bO = (long)Tk * D_; p.bI = DQ_; p.ldb = D_;
      p.C = PF(o_logits); p.cO = (long)H_ * Tq * Tk; p.cI = (long)Tq * Tk; p.ldc = Tk;
      p.M = Tq; p.N = Tk; p.K = DQ_; p.alpha = rsq; gemmL(p); }
    if (relf) {  // Shaw relative-position logits rl[b,h,t,r]
      GemmP p = gp0(); p.batches = B_ * H_; p.innerCnt = H_;
      p.A = PH(o_q); p.aO = (long)Tq * D_; p.aI = DQ_; p.lda = D_;
      p.B = PH(o_rel); p.bO = 0; p.bI = (long)RW_ * DQ_; p.ldb = DQ_;
      p.C = PF(o_rl); p.cO = (long)H_ * Tq * RW_; p.cI = (long)Tq * RW_; p.ldc = RW_;
      p.M = Tq; p.N = RW_; p.K = DQ_; p.alpha = rsq; gemmL(p);
    }
    k_softmax<<<dim3((unsigned)(B_ * H_ * Tq)), dim3(256), 0, stream>>>(
        PF(o_logits), relf ? PF(o_rl) : nullptr, pad, PH(o_probs), Tq, Tk, causal, RW_);
    // o[b,t,h,a] = probs @ v
    { GemmP p = gp0(); p.batches = B_ * H_; p.innerCnt = H_;
      p.A = PH(o_probs); p.aO = (long)H_ * Tq * Tk; p.aI = (long)Tq * Tk; p.lda = Tk;
      p.B = PH(o_vt);    p.bO = (long)H_ * DQ_ * Tk; p.bI = (long)DQ_ * Tk; p.ldb = Tk;
      p.Ch = PH(o_oa); p.hO = (long)Tq * D_; p.hI = DQ_; p.ldch = D_;
      p.M = Tq; p.N = DQ_; p.K = Tk; gemmL(p); }
    // output projection
    { GemmP p = gp0(); p.A = PH(o_oa); p.lda = D_; p.B = PH(o_wo); p.ldb = D_;
      p.C = attnout; p.ldc = D_; p.M = B_ * Tq; p.N = D_; p.K = D_; gemmL(p); }
  };

  auto ffn = [&](const _Float16* xin, int M, const float* l1w, const float* l1b,
                 const float* l2w, const float* l2b, float* outf) {
    packTL(l1w, PH(o_wl1), D_, FF_, 1);
    packTL(l2w, PH(o_wl2), FF_, D_, 1);
    { GemmP p = gp0(); p.A = xin; p.lda = D_; p.B = PH(o_wl1); p.ldb = D_;
      p.Ch = PH(o_ffh); p.ldch = FF_; p.M = M; p.N = FF_; p.K = D_;
      p.bias = l1b; p.relu = 1; gemmL(p); }
    { GemmP p = gp0(); p.A = PH(o_ffh); p.lda = FF_; p.B = PH(o_wl2); p.ldb = FF_;
      p.C = outf; p.ldc = D_; p.M = M; p.N = D_; p.K = FF_; p.bias = l2b; gemmL(p); }
  };

  // ============================== encoder ==================================
  for (int l = 0; l < L_; ++l) {
    attn(PH(o_xh), PH(o_xh), TS_, TS_,
         F(60) + (size_t)l * H_ * D_ * DQ_, F(61) + (size_t)l * H_ * D_ * DQ_,
         F(62) + (size_t)l * H_ * D_ * DQ_, F(63) + (size_t)l * H_ * DQ_ * D_,
         F(64) + (size_t)l * H_ * RW_ * DQ_, 0, PI(o_srcpad), PF(o_attn));
    k_addln<<<dim3((unsigned)(B_ * TS_)), dim3(256), 0, stream>>>(
        PF(o_xf), PF(o_attn), F(65) + (size_t)l * D_, F(66) + (size_t)l * D_, PF(o_xf), PH(o_xh));
    ffn(PH(o_xh), B_ * TS_, F(69) + (size_t)l * D_ * FF_, F(70) + (size_t)l * FF_,
        F(71) + (size_t)l * FF_ * D_, F(72) + (size_t)l * D_, PF(o_ffo));
    k_addln<<<dim3((unsigned)(B_ * TS_)), dim3(256), 0, stream>>>(
        PF(o_xf), PF(o_ffo), F(67) + (size_t)l * D_, F(68) + (size_t)l * D_, PF(o_xf), PH(o_xh));
  }

  // ============================== decoder ==================================
  for (int l = 0; l < L_; ++l) {
    // causal self-attention
    attn(PH(o_th), PH(o_th), TT_, TT_,
         F(73) + (size_t)l * H_ * D_ * DQ_, F(74) + (size_t)l * H_ * D_ * DQ_,
         F(75) + (size_t)l * H_ * D_ * DQ_, F(76) + (size_t)l * H_ * DQ_ * D_,
         nullptr, 1, PI(o_tgtpad), PF(o_attn));
    k_addln<<<dim3((unsigned)(B_ * TT_)), dim3(256), 0, stream>>>(
        PF(o_tf), PF(o_attn), F(81) + (size_t)l * D_, F(82) + (size_t)l * D_, PF(o_tf), PH(o_th));
    // cross-attention against encoder memory
    attn(PH(o_th), PH(o_xh), TT_, TS_,
         F(77) + (size_t)l * H_ * D_ * DQ_, F(78) + (size_t)l * H_ * D_ * DQ_,
         F(79) + (size_t)l * H_ * D_ * DQ_, F(80) + (size_t)l * H_ * DQ_ * D_,
         nullptr, 0, PI(o_srcpad), PF(o_attn));
    k_addln<<<dim3((unsigned)(B_ * TT_)), dim3(256), 0, stream>>>(
        PF(o_tf), PF(o_attn), F(83) + (size_t)l * D_, F(84) + (size_t)l * D_, PF(o_tf), PH(o_th));
    // FFN
    ffn(PH(o_th), B_ * TT_, F(87) + (size_t)l * D_ * FF_, F(88) + (size_t)l * FF_,
        F(89) + (size_t)l * FF_ * D_, F(90) + (size_t)l * D_, PF(o_ffo));
    k_addln<<<dim3((unsigned)(B_ * TT_)), dim3(256), 0, stream>>>(
        PF(o_tf), PF(o_ffo), F(85) + (size_t)l * D_, F(86) + (size_t)l * D_, PF(o_tf), PH(o_th));
  }

  // ======================= output projections =============================
  float* dout = (float*)d_out;
  { GemmP p = gp0(); p.A = PH(o_xh); p.lda = D_; p.B = PH(o_waux); p.ldb = D_;
    p.C = dout; p.ldc = V_; p.M = B_ * TS_; p.N = V_; p.K = D_; p.bias = F(92); gemmL(p); }
  { GemmP p = gp0(); p.A = PH(o_th); p.lda = D_; p.B = PH(o_wout); p.ldb = D_;
    p.C = dout + (size_t)B_ * TS_ * V_; p.ldc = V_; p.M = B_ * TT_; p.N = V_; p.K = D_;
    p.bias = F(94); gemmL(p); }
}
